// Attention_55808805044825
// MI455X (gfx1250) — matrix operand order repaired and checked by reading, not compiled, not run on hardware
//
#include <hip/hip_runtime.h>

typedef __bf16 bf16_t;
typedef __attribute__((ext_vector_type(16))) __bf16 v16bf;
typedef __attribute__((ext_vector_type(8)))  __bf16 v8bf;
typedef __attribute__((ext_vector_type(8)))  float  v8f;

namespace {
constexpr int NUM_HEADS     = 32;
constexpr int NUM_KV_HEADS  = 8;
constexpr int HEAD_DIM      = 128;
constexpr int PAGE_SIZE     = 16;
constexpr int PAGES_PER_SEQ = 256;
constexpr int Q_LEN         = 1024;
constexpr int T_PAST        = PAGES_PER_SEQ * PAGE_SIZE;   // 4096
constexpr int Q_STRIDE      = NUM_HEADS * HEAD_DIM;        // 4096 floats
constexpr int KV_ROW_STRIDE = NUM_KV_HEADS * HEAD_DIM;     // 1024 floats
constexpr float SCALE       = 0.08838834764831845f;
constexpr float LOG2E       = 1.44269504088896340736f;
constexpr float SCALE_LOG2  = SCALE * LOG2E;               // scores in exp2 domain
constexpr float NEG_INF     = -3.0e38f;

constexpr int BLOCK_Q  = 64;    // q rows per workgroup = 4 waves x 16
constexpr int KV_TILE  = 32;    // kv tokens per mainloop step (= 2 pages)
constexpr int NTHREADS = 128;   // 4 waves
}

// exchange with lane^16 (wave32): ds_swizzle SWAPX16 (xor=0x10, and=0x1f)
__device__ __forceinline__ float swap16(float x) {
  return __int_as_float(__builtin_amdgcn_ds_swizzle(__float_as_int(x), 0x401f));
}

__global__ __launch_bounds__(NTHREADS)
void fa_prefill_paged(const float* __restrict__ q,
                      const float* __restrict__ knew,
                      const float* __restrict__ vnew,
                      const float* __restrict__ kv_cache,
                      const int*   __restrict__ block_table,
                      float* __restrict__ out)
{
  __shared__ bf16_t sK [KV_TILE][HEAD_DIM];   // K tile, row-major     (8 KB)
  __shared__ bf16_t sVt[HEAD_DIM][KV_TILE];   // V tile, transposed    (8 KB)

  const int qt    = blockIdx.x;
  const int h     = blockIdx.y;
  const int kvh   = h >> 2;                   // GQA: 4 q-heads per kv-head
  const int tid   = threadIdx.x;
  const int lane  = tid & 31;
  const int wave  = tid >> 5;
  const int lmod  = lane & 15;                // matrix column (q index in tile)
  const int lhalf = lane >> 4;                // which half of the K-runs
  const int q_wave_base = qt * BLOCK_Q + wave * 16;
  const int q_row = q_wave_base + lmod;

  // ---- hold Q as B-fragments (Q^T), 4 chunks of K-depth 32 -----------------
  v16bf Qf[4];
  {
    const float* qp = q + (size_t)q_row * Q_STRIDE + h * HEAD_DIM;
    #pragma unroll
    for (int c = 0; c < 4; ++c) {
      const float4* p4 = (const float4*)(qp + c * 32 + lhalf * 16);
      union { float4 v4[4]; float f[16]; } u;
      u.v4[0] = p4[0]; u.v4[1] = p4[1]; u.v4[2] = p4[2]; u.v4[3] = p4[3];
      #pragma unroll
      for (int i = 0; i < 16; ++i) Qf[c][i] = (bf16_t)u.f[i];
    }
  }

  // ---- per-thread fill descriptors ----------------------------------------
  // K fill: thread owns 1 kv row x 32 d (two 32B LDS stores)
  const int k_kvr  = tid >> 2;            // 0..31
  const int k_dseg = (tid & 3) * 32;      // 0,32,64,96
  // V fill: thread owns 8 consecutive kv rows x 4 d  -> 4x 16B transposed
  const int v_d0   = (tid & 31) * 4;      // 0..124
  const int v_kv0r = (tid >> 5) * 8;      // 0,8,16,24  (stays inside one page)

  auto src_ptr = [&](int t, int sel) -> const float* {
    if (t < T_PAST) {
      const int page = block_table[t >> 4];
      return kv_cache
           + (((size_t)page * 2 + sel) * NUM_KV_HEADS + kvh) * (PAGE_SIZE * HEAD_DIM)
           + (size_t)(t & (PAGE_SIZE - 1)) * HEAD_DIM;
    }
    const float* base = sel ? vnew : knew;
    return base + (size_t)(t - T_PAST) * KV_ROW_STRIDE + kvh * HEAD_DIM;
  };

  union F32x32 { float4 v4[8]; float f[32]; };
  F32x32 kreg, vreg;

  auto prefetch = [&](int kt) {
    const int kv0 = kt * KV_TILE;
    const float4* kp4 = (const float4*)(src_ptr(kv0 + k_kvr, 0) + k_dseg);
    #pragma unroll
    for (int i = 0; i < 8; ++i) kreg.v4[i] = kp4[i];
    const float* vb = src_ptr(kv0 + v_kv0r, 1) + v_d0;
    #pragma unroll
    for (int j = 0; j < 8; ++j) vreg.v4[j] = *(const float4*)(vb + j * HEAD_DIM);
  };

  auto store_tile = [&]() {
    #pragma unroll
    for (int half = 0; half < 2; ++half) {
      v16bf kb;
      #pragma unroll
      for (int i = 0; i < 16; ++i) kb[i] = (bf16_t)kreg.f[half * 16 + i];
      *(v16bf*)&sK[k_kvr][k_dseg + half * 16] = kb;
    }
    #pragma unroll
    for (int d = 0; d < 4; ++d) {
      v8bf vb8;                               // 8 kv values at fixed d
      #pragma unroll
      for (int j = 0; j < 8; ++j) vb8[j] = (bf16_t)vreg.f[j * 4 + d];
      *(v8bf*)&sVt[v_d0 + d][v_kv0r] = vb8;   // one ds_store_b128
    }
  };

  // ---- flash-attention state (O^T accumulators: M=d, N=q) ------------------
  v8f O[8];
  #pragma unroll
  for (int c = 0; c < 8; ++c) O[c] = (v8f)0.0f;
  float m = NEG_INF, l = 0.0f;

  const int n_tiles = (qt + 1) * (BLOCK_Q / KV_TILE);  // causal skip

  prefetch(0);

  for (int kt = 0; kt < n_tiles; ++kt) {
    const int kv0 = kt * KV_TILE;

    __syncthreads();           // previous tile fully consumed
    store_tile();
    __syncthreads();           // tile visible
    if (kt + 1 < n_tiles) prefetch(kt + 1);   // overlap VMEM with WMMA

    // ---- S^T = K * Q^T : two 16-kv subtiles, K-depth 128 in 4 WMMAs each --
    float s[2][8];
    #pragma unroll
    for (int half = 0; half < 2; ++half) {
      v8f acc = (v8f)0.0f;
      const int kvr = half * 16 + lmod;
      #pragma unroll
      for (int c = 0; c < 4; ++c) {
        const int b0 = c * 32 + lhalf * 16;
        v8bf lo = *(const v8bf*)&sK[kvr][b0];
        v8bf hi = *(const v8bf*)&sK[kvr][b0 + 8];
        v16bf Kf;
        #pragma unroll
        for (int i = 0; i < 8; ++i) { Kf[i] = lo[i]; Kf[8 + i] = hi[i]; }
        acc = __builtin_amdgcn_wmma_f32_16x16x32_bf16(
                  false, Kf, false, Qf[c], (short)0, acc, false, false);
      }
      #pragma unroll
      for (int r = 0; r < 8; ++r) s[half][r] = acc[r];
    }

    // ---- online softmax in exp2 domain ------------------------------------
    float mt = NEG_INF;
    if (kv0 + KV_TILE - 1 > q_wave_base) {
      // diagonal tile for this wave: apply causal mask (wave-uniform branch)
      #pragma unroll
      for (int half = 0; half < 2; ++half)
        #pragma unroll
        for (int r = 0; r < 8; ++r) {
          const int kvi = kv0 + half * 16 + lhalf * 8 + r;
          const float val = (kvi <= q_row) ? s[half][r] * SCALE_LOG2 : NEG_INF;
          s[half][r] = val;
          mt = fmaxf(mt, val);
        }
    } else {
      // fully-visible tile: no compares
      #pragma unroll
      for (int half = 0; half < 2; ++half)
        #pragma unroll
        for (int r = 0; r < 8; ++r) {
          const float val = s[half][r] * SCALE_LOG2;
          s[half][r] = val;
          mt = fmaxf(mt, val);
        }
    }
    mt = fmaxf(mt, swap16(mt));               // other 8 kv rows live in lane^16
    const float m_new = fmaxf(m, mt);
    const float corr  = __builtin_amdgcn_exp2f(m - m_new);
    float rs = 0.0f;
    #pragma unroll
    for (int half = 0; half < 2; ++half)
      #pragma unroll
      for (int r = 0; r < 8; ++r) {
        const float p = __builtin_amdgcn_exp2f(s[half][r] - m_new);
        s[half][r] = p;
        rs += p;
      }
    rs += swap16(rs);
    l = l * corr + rs;
    m = m_new;
    #pragma unroll
    for (int c = 0; c < 8; ++c) O[c] = O[c] * corr;

    // ---- build P^T B-fragment in registers via lane^16 swaps --------------
    float e0[8], e1[8];
    #pragma unroll
    for (int r = 0; r < 8; ++r) e0[r] = swap16(s[0][r]);
    #pragma unroll
    for (int r = 0; r < 8; ++r) e1[r] = swap16(s[1][r]);
    v16bf Pf;
    #pragma unroll
    for (int i = 0; i < 8; ++i) {
      Pf[i]     = (bf16_t)(lhalf ? e1[i] : s[0][i]);
      Pf[8 + i] = (bf16_t)(lhalf ? s[1][i] : e0[i]);
    }

    // ---- O^T += V^T * P^T : 8 d-chunks, K-depth 32, one WMMA each ---------
    #pragma unroll
    for (int c = 0; c < 8; ++c) {
      const int dr = c * 16 + lmod;
      const int b0 = lhalf * 16;
      v8bf lo = *(const v8bf*)&sVt[dr][b0];
      v8bf hi = *(const v8bf*)&sVt[dr][b0 + 8];
      v16bf Vf;
      #pragma unroll
      for (int i = 0; i < 8; ++i) { Vf[i] = lo[i]; Vf[8 + i] = hi[i]; }
      O[c] = __builtin_amdgcn_wmma_f32_16x16x32_bf16(
                 false, Vf, false, Pf, (short)0, O[c], false, false);
    }
  }

  // ---- normalize and store: lane owns column q, rows d = c*16+lhalf*8+r ----
  const float inv_l = 1.0f / l;
  float* op_base = out + (size_t)q_row * Q_STRIDE + h * HEAD_DIM;
  #pragma unroll
  for (int c = 0; c < 8; ++c) {
    float4 a, b;
    a.x = O[c][0] * inv_l; a.y = O[c][1] * inv_l;
    a.z = O[c][2] * inv_l; a.w = O[c][3] * inv_l;
    b.x = O[c][4] * inv_l; b.y = O[c][5] * inv_l;
    b.z = O[c][6] * inv_l; b.w = O[c][7] * inv_l;
    float* op = op_base + c * 16 + lhalf * 8;
    ((float4*)op)[0] = a;
    ((float4*)op)[1] = b;
  }
}

extern "C" void kernel_launch(void* const* d_in, const int* in_sizes, int n_in,
                              void* d_out, int out_size, void* d_ws, size_t ws_size,
                              hipStream_t stream) {
  const float* q        = (const float*)d_in[0];
  const float* k        = (const float*)d_in[1];
  const float* v        = (const float*)d_in[2];
  const float* kv_cache = (const float*)d_in[3];
  const int*   bt       = (const int*)d_in[4];
  float* out            = (float*)d_out;

  dim3 grid(Q_LEN / BLOCK_Q, NUM_HEADS);   // 16 x 32 = 512 workgroups
  fa_prefill_paged<<<grid, dim3(NTHREADS), 0, stream>>>(q, k, v, kv_cache, bt, out);

  (void)in_sizes; (void)n_in; (void)out_size; (void)d_ws; (void)ws_size;
}
